// QuanvolutionClassifier_65481071408733
// MI455X (gfx1250) — matrix operand order repaired and checked by reading, not compiled, not run on hardware
//
#include <hip/hip_runtime.h>
#include <hip/hip_bf16.h>
#include <math.h>

typedef __attribute__((ext_vector_type(16))) _Float16 v16h;
typedef __attribute__((ext_vector_type(2)))  _Float16 v2h;
typedef __attribute__((ext_vector_type(8)))  float    v8f;

#define QSTEPS 8            // quantum GEMM: K = 256 = 8 * 32
#define CSTEPS 25           // classifier GEMM: K = 800 = 25 * 32 (784 padded)
#define CK     800
#define IMGS_PER_BLOCK 16
#define PPI    196          // patches per image
#define BQ_OFF 0
#define WP_OFF (QSTEPS*32*16)   // in _Float16 units

union V16 { v16h v; v2h p[8]; };

struct C2 { float r, i; };
__device__ __forceinline__ C2 cmul(C2 a, C2 b) {
  return { a.r*b.r - a.i*b.i, a.r*b.i + a.i*b.r };
}

// ---------------------------------------------------------------------------
// Precompute kernel (1 block): build M = P2 U2 P1 U1, A_w, then pack the
// quantum B-operand (256x16, f16) and classifier W B-operand (800x16, f16)
// directly in WMMA per-lane register layout into the workspace.
// ---------------------------------------------------------------------------
__global__ void quanv_precompute(const float* __restrict__ weights,
                                 const float* __restrict__ W,
                                 _Float16* __restrict__ ws)
{
  __shared__ float rotr[2][4][2][2], roti[2][4][2][2];
  __shared__ int   perm[2][16];
  __shared__ float Ur[2][16][16], Ui[2][16][16];
  __shared__ float Mar[16][16], Mai[16][16];
  __shared__ float Tr[16][16],  Ti[16][16];
  __shared__ float Mbr[16][16], Mbi[16][16];
  __shared__ float Amat[4][16][16];

  const int tid = threadIdx.x;

  if (tid == 0) {
    for (int l = 0; l < 2; ++l)
      for (int q = 0; q < 4; ++q) {
        float phi   = weights[(l*4+q)*3 + 0];
        float theta = weights[(l*4+q)*3 + 1];
        float omega = weights[(l*4+q)*3 + 2];
        float ct = __cosf(0.5f*theta), st = __sinf(0.5f*theta);
        float ap = 0.5f*(phi+omega), am = 0.5f*(phi-omega);
        C2 ep = { __cosf(ap), -__sinf(ap) };   // exp(-i(phi+omega)/2)
        C2 em = { __cosf(am),  __sinf(am) };   // exp(+i(phi-omega)/2)
        rotr[l][q][0][0] =  ep.r*ct; roti[l][q][0][0] =  ep.i*ct;  // ep*ct
        rotr[l][q][0][1] = -em.r*st; roti[l][q][0][1] = -em.i*st;  // -em*st
        rotr[l][q][1][0] =  em.r*st; roti[l][q][1][0] = -em.i*st;  // conj(em)*st
        rotr[l][q][1][1] =  ep.r*ct; roti[l][q][1][1] = -ep.i*ct;  // conj(ep)*ct
      }
    for (int l = 0; l < 2; ++l) {
      int r = l % 3 + 1;
      int p[16], np[16];
      for (int k = 0; k < 16; ++k) p[k] = k;
      for (int i = 0; i < 4; ++i) {
        int cbit = 3 - i;
        int tbit = 3 - ((i + r) & 3);
        for (int k = 0; k < 16; ++k) {
          int cm = k ^ (((k >> cbit) & 1) << tbit);
          np[k] = p[cm];
        }
        for (int k = 0; k < 16; ++k) p[k] = np[k];
      }
      for (int k = 0; k < 16; ++k) perm[l][k] = p[k];
    }
  }
  __syncthreads();

  {  // U_l = kron(rot_l0, rot_l1, rot_l2, rot_l3)
    int r = tid >> 4, c = tid & 15;
    for (int l = 0; l < 2; ++l) {
      C2 acc = { 1.f, 0.f };
      #pragma unroll
      for (int q = 0; q < 4; ++q) {
        int br = (r >> (3-q)) & 1, bc = (c >> (3-q)) & 1;
        C2 rq = { rotr[l][q][br][bc], roti[l][q][br][bc] };
        acc = cmul(acc, rq);
      }
      Ur[l][r][c] = acc.r; Ui[l][r][c] = acc.i;
    }
  }
  __syncthreads();
  {  // Ma = rowgather(U0, perm0)
    int k = tid >> 4, j = tid & 15, pk = perm[0][k];
    Mar[k][j] = Ur[0][pk][j]; Mai[k][j] = Ui[0][pk][j];
  }
  __syncthreads();
  {  // T = U1 @ Ma
    int k = tid >> 4, j = tid & 15;
    C2 acc = { 0.f, 0.f };
    for (int m = 0; m < 16; ++m) {
      C2 a = { Ur[1][k][m], Ui[1][k][m] };
      C2 b = { Mar[m][j],   Mai[m][j]   };
      C2 p = cmul(a, b);
      acc.r += p.r; acc.i += p.i;
    }
    Tr[k][j] = acc.r; Ti[k][j] = acc.i;
  }
  __syncthreads();
  {  // Mb = rowgather(T, perm1)
    int k = tid >> 4, j = tid & 15, pk = perm[1][k];
    Mbr[k][j] = Tr[pk][j]; Mbi[k][j] = Ti[pk][j];
  }
  __syncthreads();
  // A_w[i][j] = sum_k Z[k][w] * (Re M[k][i] Re M[k][j] + Im M[k][i] Im M[k][j])
  for (int e = tid; e < 4*256; e += 256) {
    int w = e >> 8, i = (e >> 4) & 15, j = e & 15;
    float acc = 0.f;
    for (int k = 0; k < 16; ++k) {
      float z = ((k >> (3-w)) & 1) ? -1.f : 1.f;
      acc += z * (Mbr[k][i]*Mbr[k][j] + Mbi[k][i]*Mbi[k][j]);
    }
    Amat[w][i][j] = acc;
  }
  __syncthreads();
  // Quantum B operand pack: B[k][n] = A_n[k>>4][k&15], k = 32m + 16*(j>>3) + 8*(L>>4) + (j&7)
  for (int f = tid; f < QSTEPS*32*16; f += 256) {
    int m = f >> 9, L = (f >> 4) & 31, j = f & 15;
    int k = m*32 + ((j >> 3) << 4) + ((L >> 4) << 3) + (j & 7);
    int c = L & 15;
    float v = (c < 4) ? Amat[c][k >> 4][k & 15] : 0.f;
    ws[BQ_OFF + f] = (_Float16)v;
  }
  // Classifier B operand pack: B[k][n] = W[n][k], zero-padded (n>=10, k>=784)
  for (int f = tid; f < CSTEPS*32*16; f += 256) {
    int m = f >> 9, L = (f >> 4) & 31, j = f & 15;
    int k = m*32 + ((j >> 3) << 4) + ((L >> 4) << 3) + (j & 7);
    int c = L & 15;
    float v = (c < 10 && k < 784) ? W[c*784 + k] : 0.f;
    ws[WP_OFF + f] = (_Float16)v;
  }
}

// ---------------------------------------------------------------------------
// Main kernel: one block = 16 images (8 waves). Each wave handles 16-patch
// tiles; per tile: 8x v_wmma_f32_16x16x32_f16 with the A operand built via
// packed f16 outer-product (v_pk_mul_f16 writes operand VGPRs directly).
// Features staged as f16 in LDS; wave 0 runs the 16x800x16 classifier WMMA
// chain + bias + log_softmax.
// ---------------------------------------------------------------------------
__global__ void __launch_bounds__(256)
quanv_main(const float* __restrict__ x,
           const float* __restrict__ bias,
           const _Float16* __restrict__ ws,
           float* __restrict__ out)
{
  __shared__ _Float16 feat[IMGS_PER_BLOCK][CK];  // 16 x 800 f16 (784 + pad)
  __shared__ float    lg[16][16];

  const int tid  = threadIdx.x;
  const int lane = tid & 31;
  const int wid  = tid >> 5;
  const int h    = lane >> 4;     // lane half: selects K sub-slice
  const int c16  = lane & 15;

  // zero the K padding columns once
  feat[tid >> 4][784 + (tid & 15)] = (_Float16)0.f;

  // constant quantum B operand: one 32B vector per lane per k-step
  v16h bq[QSTEPS];
  #pragma unroll
  for (int m = 0; m < QSTEPS; ++m)
    bq[m] = *(const v16h*)(ws + BQ_OFF + (m*32 + lane)*16);

  const size_t img_base = (size_t)blockIdx.x * IMGS_PER_BLOCK;

  for (int t = wid; t < PPI; t += 8) {          // 196 tiles of 16 patches
    // A-operand row = patch (lane&15); this lane's pixels:
    int lp  = t*16 + c16;
    int il  = lp / PPI;
    int pid = lp - il*PPI;
    int py  = pid / 14;
    int px  = pid - py*14;
    const float* ip = x + (img_base + il)*784 + py*56 + px*2;
    float p0 = ip[0], p1 = ip[1], p2 = ip[28], p3 = ip[29];

    float c0 = __cosf(0.5f*p0), s0 = __sinf(0.5f*p0);
    float c1 = __cosf(0.5f*p1), s1 = __sinf(0.5f*p1);
    float c2 = __cosf(0.5f*p2), s2 = __sinf(0.5f*p2);
    float c3 = __cosf(0.5f*p3), s3 = __sinf(0.5f*p3);
    float t00 = c2*c3, t01 = c2*s3, t10 = s2*c3, t11 = s2*s3;
    float a00 = c0*c1, a01 = c0*s1, a10 = s0*c1, a11 = s0*s1;
    float sv[16];
    sv[ 0]=a00*t00; sv[ 1]=a00*t01; sv[ 2]=a00*t10; sv[ 3]=a00*t11;
    sv[ 4]=a01*t00; sv[ 5]=a01*t01; sv[ 6]=a01*t10; sv[ 7]=a01*t11;
    sv[ 8]=a10*t00; sv[ 9]=a10*t01; sv[10]=a10*t10; sv[11]=a10*t11;
    sv[12]=a11*t00; sv[13]=a11*t01; sv[14]=a11*t10; sv[15]=a11*t11;

    // i-side: broadcast-packed f16 (v_cvt_pk_f16_f32 vD, s, s)
    v2h svbc[16];
    #pragma unroll
    for (int i = 0; i < 16; ++i) {
      _Float16 q = (_Float16)sv[i];
      svbc[i] = (v2h){ q, q };
    }
    // j-side: this lane's K sub-slice, packed in sequential pairs
    v2h sbp[4];
    #pragma unroll
    for (int t2 = 0; t2 < 4; ++t2) {
      float lo = h ? sv[8 + 2*t2]     : sv[2*t2];
      float hi = h ? sv[8 + 2*t2 + 1] : sv[2*t2 + 1];
      sbp[t2] = (v2h){ (_Float16)lo, (_Float16)hi };
    }

    v8f acc = {};
    #pragma unroll
    for (int m = 0; m < QSTEPS; ++m) {
      // u[k] = s[k>>4]*s[k&15]; k = 32m + 16*(a-half) + 8h + jj -> i = 2m(+1)
      V16 A;
      #pragma unroll
      for (int t2 = 0; t2 < 4; ++t2) {
        A.p[t2]     = svbc[2*m]   * sbp[t2];   // v_pk_mul_f16
        A.p[t2 + 4] = svbc[2*m+1] * sbp[t2];
      }
      acc = __builtin_amdgcn_wmma_f32_16x16x32_f16(
          false, A.v, false, bq[m], (short)0, acc, false, false);
    }

    // D element (M = r+8h, N = c16): N = wire for c16 < 4
    if (c16 < 4) {
      int base = t*16 + 8*h;
      int il0  = base / PPI;
      int rem0 = base - il0*PPI;
      #pragma unroll
      for (int r = 0; r < 8; ++r) {
        int rem   = rem0 + r;
        int cross = (rem >= PPI) ? 1 : 0;
        int ill   = il0 + cross;
        int pidd  = rem - (cross ? PPI : 0);
        feat[ill][pidd*4 + c16] = (_Float16)acc[r];
      }
    }
  }
  __syncthreads();

  // ------ classifier: wave 0 does (16 imgs x 800) x (800 x 16 classes) -----
  if (tid < 32) {
    v8f acc = {};
    for (int m = 0; m < CSTEPS; ++m) {
      v16h wv = *(const v16h*)(ws + WP_OFF + (m*32 + lane)*16);
      V16 A;
      int kb = m*32 + h*8;                         // even -> 16B aligned
      const v2h* fp  = (const v2h*)&feat[c16][kb];
      const v2h* fp2 = (const v2h*)&feat[c16][kb + 16];
      #pragma unroll
      for (int t2 = 0; t2 < 4; ++t2) {
        A.p[t2]     = fp[t2];
        A.p[t2 + 4] = fp2[t2];
      }
      acc = __builtin_amdgcn_wmma_f32_16x16x32_f16(
          false, A.v, false, wv, (short)0, acc, false, false);
    }
    float bv = (c16 < 10) ? bias[c16] : 0.f;
    #pragma unroll
    for (int r = 0; r < 8; ++r)
      lg[r + 8*h][c16] = acc[r] + bv;     // lg[img][class]
  }
  __builtin_amdgcn_wave_barrier();
  if (tid < 16) {
    float v[10];
    float mx = -1e30f;
    #pragma unroll
    for (int c = 0; c < 10; ++c) { v[c] = lg[tid][c]; mx = fmaxf(mx, v[c]); }
    float ssum = 0.f;
    #pragma unroll
    for (int c = 0; c < 10; ++c) ssum += __expf(v[c] - mx);
    float lse = mx + __logf(ssum);
    float* op = out + (img_base + tid)*10;
    #pragma unroll
    for (int c = 0; c < 10; ++c) op[c] = v[c] - lse;
  }
}

extern "C" void kernel_launch(void* const* d_in, const int* in_sizes, int n_in,
                              void* d_out, int out_size, void* d_ws, size_t ws_size,
                              hipStream_t stream) {
  const float* x  = (const float*)d_in[0];   // (8192,1,28,28)
  const float* wt = (const float*)d_in[1];   // (2,4,3)
  const float* W  = (const float*)d_in[2];   // (10,784)
  const float* bb = (const float*)d_in[3];   // (10,)
  float* out = (float*)d_out;
  _Float16* ws = (_Float16*)d_ws;

  int nimg = in_sizes[0] / 784;
  int grid = nimg / IMGS_PER_BLOCK;

  quanv_precompute<<<1, 256, 0, stream>>>(wt, W, ws);
  quanv_main<<<grid, 256, 0, stream>>>(x, bb, (const _Float16*)ws, out);
}
